// iDecoderJamba_1984274891166
// MI455X (gfx1250) — matrix operand order repaired and checked by reading, not compiled, not run on hardware
//
#include <hip/hip_runtime.h>
#include <hip/hip_bf16.h>

// ---------------------------------------------------------------------------
// Types for CDNA5 WMMA (wave32): bf16 A/B fragments (16 x bf16), f32 C/D (8 x f32)
// ---------------------------------------------------------------------------
typedef __attribute__((ext_vector_type(16))) __bf16 v16bf;
typedef __attribute__((ext_vector_type(8)))  __bf16 v8bf;
typedef __attribute__((ext_vector_type(4)))  __bf16 v4bf;
typedef __attribute__((ext_vector_type(8)))  float  v8f;

#if __has_builtin(__builtin_amdgcn_global_load_async_to_lds_b32) && \
    __has_builtin(__builtin_amdgcn_s_wait_asynccnt)
#define SCAN_ASYNC 1
#endif

typedef __attribute__((address_space(1))) int gint_t;
typedef __attribute__((address_space(3))) int lint_t;

__device__ __forceinline__ __bf16 f2bf(float f) {
    unsigned u = __builtin_bit_cast(unsigned, f);
    unsigned r = u + 0x7FFFu + ((u >> 16) & 1u);   // round-to-nearest-even
    unsigned short hs = (unsigned short)(r >> 16);
    return __builtin_bit_cast(__bf16, hs);
}

__device__ __forceinline__ float sigmoidf_(float x) { return 1.f / (1.f + __expf(-x)); }

// ---------------------------------------------------------------------------
// GEMM: C[M,N] = act(A[M,K] @ W[K,N] + bias) + resid
//   A row stride lda, W row stride ldw (cols 0..N-1 used), C/resid stride ldc.
//   M % 64 == 0, K % 32 == 0. N arbitrary (clamped loads + guarded stores).
//
// Block = 256 threads = 8 waves (2x4). Block tile 64(M) x 128(N).
// Each wave: 32x32 tile = 2x2 fragments -> 4 v_wmma per K-step.
// K-step (32) pipeline: regs(k+1) global-loads issued before compute(k).
//   A LDS: row-major  [64][40] bf16 ; B LDS: transposed [128][40] bf16.
// act: 0=none, 1=silu, 2=softplus
// ---------------------------------------------------------------------------
__global__ __launch_bounds__(256) void gemm_wmma_kernel(
    const float* __restrict__ A, int lda,
    const float* __restrict__ W, int ldw,
    const float* __restrict__ bias,
    const float* __restrict__ resid,
    float* __restrict__ C, int ldc,
    int M, int N, int K, int act)
{
    __shared__ __bf16 Alds[64 * 40];
    __shared__ __bf16 Blds[128 * 40];

    const int t    = threadIdx.x;
    const int wave = t >> 5;
    const int lane = t & 31;
    const int wm   = wave >> 2;            // 0..1 : 32-row slab
    const int wn   = wave & 3;             // 0..3 : 32-col slab
    const int half = lane >> 4;            // 0/1
    const int mr   = lane & 15;

    const int rowblk = blockIdx.y * 64;
    const int colblk = blockIdx.x * 128;

    v8f acc[2][2];
    #pragma unroll
    for (int r = 0; r < 2; r++)
        #pragma unroll
        for (int c = 0; c < 2; c++)
            acc[r][c] = (v8f){0.f,0.f,0.f,0.f,0.f,0.f,0.f,0.f};

    // staging index maps (constant per thread)
    const int ar  = t >> 3;                // A row 0..31 (+32 second pass)
    const int ac4 = (t & 7) << 2;          // A col 0,4,..,28
    const int br  = t >> 5;                // B row 0..7  (+8/16/24)
    const int bc4 = (t & 31) << 2;         // B col 0,4,..,124
    int bcol = colblk + bc4;
    if (bcol > N - 4) bcol = N - 4;        // clamp; extra cols never stored

    float4 areg[2];
    float  breg[4][4];

    auto load_stage = [&](int kk) {
        #pragma unroll
        for (int ii = 0; ii < 2; ii++)
            areg[ii] = *(const float4*)(A + (size_t)(rowblk + ar + ii * 32) * lda + kk + ac4);
        #pragma unroll
        for (int ii = 0; ii < 4; ii++) {
            const float* src = W + (size_t)(kk + br + ii * 8) * ldw + bcol;
            breg[ii][0] = src[0]; breg[ii][1] = src[1];
            breg[ii][2] = src[2]; breg[ii][3] = src[3];
        }
    };
    auto store_stage = [&]() {
        #pragma unroll
        for (int ii = 0; ii < 2; ii++) {
            v4bf pk;
            pk[0] = f2bf(areg[ii].x); pk[1] = f2bf(areg[ii].y);
            pk[2] = f2bf(areg[ii].z); pk[3] = f2bf(areg[ii].w);
            *(v4bf*)(Alds + (ar + ii * 32) * 40 + ac4) = pk;
        }
        #pragma unroll
        for (int ii = 0; ii < 4; ii++) {
            int r = br + ii * 8;
            Blds[(bc4 + 0) * 40 + r] = f2bf(breg[ii][0]);
            Blds[(bc4 + 1) * 40 + r] = f2bf(breg[ii][1]);
            Blds[(bc4 + 2) * 40 + r] = f2bf(breg[ii][2]);
            Blds[(bc4 + 3) * 40 + r] = f2bf(breg[ii][3]);
        }
    };

    load_stage(0);
    for (int kk = 0; kk < K; kk += 32) {
        __syncthreads();                    // readers done with previous tiles
        store_stage();
        __syncthreads();                    // tiles published
        if (kk + 32 < K) load_stage(kk + 32);   // overlap next loads with compute
        if (kk + 64 < K) {                       // distance-2 cache prefetch
            __builtin_prefetch(A + (size_t)(rowblk + ar) * lda + kk + 64 + ac4, 0, 1);
            __builtin_prefetch(W + (size_t)(kk + 64 + br) * ldw + bcol, 0, 1);
        }

        // ---- fragments from LDS ----
        v16bf afrag[2], bfrag[2];
        #pragma unroll
        for (int r = 0; r < 2; r++) {
            // A layout: a[0..7] = K 8*half..+7 ; a[8..15] = K 16+8*half..+7
            const __bf16* base = Alds + (wm * 32 + r * 16 + mr) * 40;
            v8bf lo = *(const v8bf*)(base + 8 * half);
            v8bf hi = *(const v8bf*)(base + 16 + 8 * half);
            #pragma unroll
            for (int j = 0; j < 8; j++) { afrag[r][j] = lo[j]; afrag[r][8 + j] = hi[j]; }
        }
        #pragma unroll
        for (int c = 0; c < 2; c++) {
            // B layout: b[0..7] = K 8*half..+7 ; b[8..15] = K 16+8*half..+7
            const __bf16* base = Blds + (wn * 32 + c * 16 + mr) * 40;
            v8bf lo = *(const v8bf*)(base + 8 * half);
            v8bf hi = *(const v8bf*)(base + 16 + 8 * half);
            #pragma unroll
            for (int j = 0; j < 8; j++) { bfrag[c][j] = lo[j]; bfrag[c][8 + j] = hi[j]; }
        }

        #pragma unroll
        for (int r = 0; r < 2; r++)
            #pragma unroll
            for (int c = 0; c < 2; c++)
                acc[r][c] = __builtin_amdgcn_wmma_f32_16x16x32_bf16(
                    false, afrag[r], false, bfrag[c],
                    (short)0, acc[r][c], false, false);
    }

    // ---- epilogue: C/D layout -> M = j + 8*half, N = lane&15 ----
    #pragma unroll
    for (int r = 0; r < 2; r++) {
        #pragma unroll
        for (int c = 0; c < 2; c++) {
            int col = colblk + wn * 32 + c * 16 + mr;
            if (col < N) {
                #pragma unroll
                for (int j = 0; j < 8; j++) {
                    int row = rowblk + wm * 32 + r * 16 + j + 8 * half;
                    float v = acc[r][c][j];
                    if (bias)  v += bias[col];
                    if (act == 1)      v = v * sigmoidf_(v);
                    else if (act == 2) v = (v > 20.f) ? v : log1pf(__expf(v));
                    if (resid) v += resid[(size_t)row * ldc + col];
                    C[(size_t)row * ldc + col] = v;
                }
            }
        }
    }
}

// ---------------------------------------------------------------------------
// Attention scores via WMMA: S[b,h,l,m] = (q_l . k_m)/sqrt(HD), causal mask.
// Both operands contiguous along head-dim -> direct global fragment loads.
// Block tile 64(query) x 128(key); 8 waves, 32x32 per wave; K-loop = HD/32.
// ---------------------------------------------------------------------------
__global__ __launch_bounds__(256) void attn_score_kernel(
    const float* __restrict__ Q, const float* __restrict__ Km,
    float* __restrict__ S, int Ls)
{
    const int DM = 1024, HD = 128;
    const int bh = blockIdx.z;             // b*8 + h
    const int b  = bh >> 3;
    const int hh = bh & 7;

    const int wave = threadIdx.x >> 5;
    const int lane = threadIdx.x & 31;
    const int wm   = wave >> 2;
    const int wn   = wave & 3;
    const int half = lane >> 4;
    const int mr   = lane & 15;

    const int qblk = blockIdx.y * 64;
    const int kblk = blockIdx.x * 128;

    v8f acc[2][2];
    #pragma unroll
    for (int r = 0; r < 2; r++)
        #pragma unroll
        for (int c = 0; c < 2; c++)
            acc[r][c] = (v8f){0.f,0.f,0.f,0.f,0.f,0.f,0.f,0.f};

    for (int kk = 0; kk < HD; kk += 32) {
        v16bf af[2], bfr[2];
        #pragma unroll
        for (int r = 0; r < 2; r++) {
            int qrow = qblk + wm * 32 + r * 16 + mr;
            const float* qp = Q + (size_t)(b * Ls + qrow) * DM + hh * HD + kk;
            float4 x0 = *(const float4*)(qp + 8 * half);
            float4 x1 = *(const float4*)(qp + 8 * half + 4);
            float4 x2 = *(const float4*)(qp + 16 + 8 * half);
            float4 x3 = *(const float4*)(qp + 16 + 8 * half + 4);
            af[r][0]  = f2bf(x0.x); af[r][1]  = f2bf(x0.y); af[r][2]  = f2bf(x0.z); af[r][3]  = f2bf(x0.w);
            af[r][4]  = f2bf(x1.x); af[r][5]  = f2bf(x1.y); af[r][6]  = f2bf(x1.z); af[r][7]  = f2bf(x1.w);
            af[r][8]  = f2bf(x2.x); af[r][9]  = f2bf(x2.y); af[r][10] = f2bf(x2.z); af[r][11] = f2bf(x2.w);
            af[r][12] = f2bf(x3.x); af[r][13] = f2bf(x3.y); af[r][14] = f2bf(x3.z); af[r][15] = f2bf(x3.w);
        }
        #pragma unroll
        for (int c = 0; c < 2; c++) {
            int kcol = kblk + wn * 32 + c * 16 + mr;
            const float* kp = Km + (size_t)(b * Ls + kcol) * DM + hh * HD + kk;
            float4 y0 = *(const float4*)(kp + 8 * half);
            float4 y1 = *(const float4*)(kp + 8 * half + 4);
            float4 y2 = *(const float4*)(kp + 16 + 8 * half);
            float4 y3 = *(const float4*)(kp + 16 + 8 * half + 4);
            bfr[c][0]  = f2bf(y0.x); bfr[c][1]  = f2bf(y0.y); bfr[c][2]  = f2bf(y0.z); bfr[c][3]  = f2bf(y0.w);
            bfr[c][4]  = f2bf(y1.x); bfr[c][5]  = f2bf(y1.y); bfr[c][6]  = f2bf(y1.z); bfr[c][7]  = f2bf(y1.w);
            bfr[c][8]  = f2bf(y2.x); bfr[c][9]  = f2bf(y2.y); bfr[c][10] = f2bf(y2.z); bfr[c][11] = f2bf(y2.w);
            bfr[c][12] = f2bf(y3.x); bfr[c][13] = f2bf(y3.y); bfr[c][14] = f2bf(y3.z); bfr[c][15] = f2bf(y3.w);
        }
        #pragma unroll
        for (int r = 0; r < 2; r++)
            #pragma unroll
            for (int c = 0; c < 2; c++)
                acc[r][c] = __builtin_amdgcn_wmma_f32_16x16x32_bf16(
                    false, af[r], false, bfr[c], (short)0, acc[r][c], false, false);
    }

    const float scale = 0.08838834764831845f;   // 1/sqrt(128)
    #pragma unroll
    for (int r = 0; r < 2; r++) {
        #pragma unroll
        for (int c = 0; c < 2; c++) {
            int col = kblk + wn * 32 + c * 16 + mr;
            #pragma unroll
            for (int j = 0; j < 8; j++) {
                int row = qblk + wm * 32 + r * 16 + j + 8 * half;
                float v = acc[r][c][j] * scale;
                if (col > row) v = -1e30f;      // causal mask
                S[((size_t)bh * Ls + row) * Ls + col] = v;
            }
        }
    }
}

// ---------------------------------------------------------------------------
// In-place row softmax over rows of length Ls
// ---------------------------------------------------------------------------
__global__ void softmax_row_kernel(float* __restrict__ S, int Ls)
{
    float* r = S + (size_t)blockIdx.x * Ls;
    __shared__ float red[256];
    int tid = threadIdx.x;

    float lm = -3.0e38f;
    for (int i = tid; i < Ls; i += 256) lm = fmaxf(lm, r[i]);
    red[tid] = lm; __syncthreads();
    for (int st = 128; st > 0; st >>= 1) {
        if (tid < st) red[tid] = fmaxf(red[tid], red[tid + st]);
        __syncthreads();
    }
    float mx = red[0];
    __syncthreads();

    float ls = 0.f;
    for (int i = tid; i < Ls; i += 256) { float p = __expf(r[i] - mx); r[i] = p; ls += p; }
    red[tid] = ls; __syncthreads();
    for (int st = 128; st > 0; st >>= 1) {
        if (tid < st) red[tid] += red[tid + st];
        __syncthreads();
    }
    float inv = 1.f / red[0];
    for (int i = tid; i < Ls; i += 256) r[i] *= inv;
}

// ---------------------------------------------------------------------------
// RMSNorm over rows of length d
// ---------------------------------------------------------------------------
__global__ void rmsnorm_kernel(const float* __restrict__ x,
                               const float* __restrict__ w,
                               float* __restrict__ out, int d)
{
    const int row = blockIdx.x;
    const float* xr = x + (size_t)row * d;
    float s = 0.f;
    for (int i = threadIdx.x; i < d; i += blockDim.x) { float v = xr[i]; s += v * v; }
    __shared__ float red[256];
    red[threadIdx.x] = s;
    __syncthreads();
    for (int st = 128; st > 0; st >>= 1) {
        if (threadIdx.x < st) red[threadIdx.x] += red[threadIdx.x + st];
        __syncthreads();
    }
    float inv = rsqrtf(red[0] / (float)d + 1e-6f);
    for (int i = threadIdx.x; i < d; i += blockDim.x)
        out[(size_t)row * d + i] = xr[i] * inv * w[i];
}

// ---------------------------------------------------------------------------
// Per-token RMSNorm of ssm rows [ntok, 96] -> dt(0:64)/B(64:80)/C(80:96), in place
// ---------------------------------------------------------------------------
__global__ void ssm_norm_kernel(float* __restrict__ ssm,
                                const float* __restrict__ dt_ln,
                                const float* __restrict__ b_ln,
                                const float* __restrict__ c_ln, int ntok)
{
    int t = blockIdx.x * blockDim.x + threadIdx.x;
    if (t >= ntok) return;
    float* r = ssm + (size_t)t * 96;
    float s = 0.f;
    for (int i = 0; i < 64; i++) { float v = r[i]; s += v * v; }
    float inv = rsqrtf(s / 64.f + 1e-6f);
    for (int i = 0; i < 64; i++) r[i] = r[i] * inv * dt_ln[i];
    s = 0.f;
    for (int i = 64; i < 80; i++) { float v = r[i]; s += v * v; }
    inv = rsqrtf(s / 16.f + 1e-6f);
    for (int i = 64; i < 80; i++) r[i] = r[i] * inv * b_ln[i - 64];
    s = 0.f;
    for (int i = 80; i < 96; i++) { float v = r[i]; s += v * v; }
    inv = rsqrtf(s / 16.f + 1e-6f);
    for (int i = 80; i < 96; i++) r[i] = r[i] * inv * c_ln[i - 80];
}

// ---------------------------------------------------------------------------
// Causal depthwise conv (D_CONV=4) over u = xz[..., :2048], then SiLU.
// ---------------------------------------------------------------------------
__global__ void conv_silu_kernel(const float* __restrict__ xz,
                                 const float* __restrict__ cw,
                                 const float* __restrict__ cb,
                                 float* __restrict__ uo, int Ls, int ntok)
{
    const int Di = 2048;
    long long idx = (long long)blockIdx.x * blockDim.x + threadIdx.x;
    if (idx >= (long long)ntok * Di) return;
    int d   = (int)(idx % Di);
    int tok = (int)(idx / Di);
    int l   = tok % Ls;
    float acc = cb[d];
    #pragma unroll
    for (int i = 0; i < 4; i++) {
        int li = l - 3 + i;
        if (li >= 0) acc += xz[(size_t)(tok - 3 + i) * 4096 + d] * cw[i * Di + d];
    }
    uo[idx] = acc * sigmoidf_(acc);
}

// ---------------------------------------------------------------------------
// Selective-scan: one thread per (b, d); h[16] in registers.
// B_t/C_t double-buffered in LDS; next step's 32 floats are brought in with
// GLOBAL_LOAD_ASYNC_TO_LDS_B32 (ASYNCcnt) when the toolchain exposes it.
// Fused  y = (scan + u*Dp) * silu(z).
// ---------------------------------------------------------------------------
__global__ void mamba_scan_kernel(const float* __restrict__ dt,
                                  const float* __restrict__ ssm,
                                  const float* __restrict__ u,
                                  const float* __restrict__ xz,
                                  const float* __restrict__ A_log,
                                  const float* __restrict__ Dp,
                                  float* __restrict__ y, int Ls)
{
    const int Di = 2048;
    const int bpb = Di / 256;
    int b = blockIdx.x / bpb;
    int d = (blockIdx.x % bpb) * 256 + threadIdx.x;

    float A[16], h[16];
    #pragma unroll
    for (int n = 0; n < 16; n++) { A[n] = -__expf(A_log[d * 16 + n]); h[n] = 0.f; }
    float Dv = Dp[d];

    __shared__ float bc[2][32];             // [buf][0:16]=B, [16:32]=C

    auto issue_bc = [&](int buf, size_t tok) {
        if (threadIdx.x < 32) {
#ifdef SCAN_ASYNC
            __builtin_amdgcn_global_load_async_to_lds_b32(
                (gint_t*)(ssm + tok * 96 + 64 + threadIdx.x),
                (lint_t*)&bc[buf][threadIdx.x],
                0, 0);
#else
            bc[buf][threadIdx.x] = ssm[tok * 96 + 64 + threadIdx.x];
#endif
        }
    };

    size_t tok0 = (size_t)b * Ls;
    issue_bc(0, tok0);
    for (int tt = 0; tt < Ls; tt++) {
        int cur = tt & 1;
        size_t tok = tok0 + tt;
#ifdef SCAN_ASYNC
        __builtin_amdgcn_s_wait_asynccnt(0);
#endif
        __syncthreads();                    // bc[cur] ready & visible to all
        if (tt + 1 < Ls) issue_bc(1 - cur, tok + 1);

        float dtv = dt[tok * Di + d];
        float uv  = u[tok * Di + d];
        float acc = 0.f;
        #pragma unroll
        for (int n = 0; n < 16; n++) {
            h[n] = __expf(dtv * A[n]) * h[n] + dtv * bc[cur][n] * uv;
            acc += h[n] * bc[cur][16 + n];
        }
        acc += uv * Dv;
        float zv = xz[tok * 4096 + 2048 + d];
        y[tok * Di + d] = acc * (zv * sigmoidf_(zv));
    }
}

__global__ void mul_kernel(float* __restrict__ a, const float* __restrict__ b, long long n)
{
    long long i = (long long)blockIdx.x * blockDim.x + threadIdx.x;
    if (i < n) a[i] *= b[i];
}

// ---------------------------------------------------------------------------
// Host orchestration
// ---------------------------------------------------------------------------
static inline void launch_gemm(const float* A, int lda, const float* W, int ldw,
                               const float* bias, const float* resid,
                               float* C, int ldc, int M, int N, int K, int act,
                               hipStream_t s)
{
    dim3 grid((N + 127) / 128, (M + 63) / 64);
    gemm_wmma_kernel<<<grid, 256, 0, s>>>(A, lda, W, ldw, bias, resid, C, ldc, M, N, K, act);
}

extern "C" void kernel_launch(void* const* d_in, const int* in_sizes, int n_in,
                              void* d_out, int out_size, void* d_ws, size_t ws_size,
                              hipStream_t stream)
{
    (void)in_sizes; (void)n_in; (void)out_size; (void)ws_size;
    const int Bb = 2, Ls = 1024, D = 1024, DFF = 4096, DI = 2048, NV = 32002;
    const int H = 8, HD = 128;
    const int M = Bb * Ls;                          // 2048 tokens

    struct LayerP {
        const float *ln1, *ln2, *gate, *up, *down;
        const float *wq, *wk, *wv, *wo;             // attn (layer 4)
        const float *in_proj, *conv_w, *conv_b, *x_proj;
        const float *dt_ln, *b_ln, *c_ln, *dt_w, *dt_b, *A_log, *Dp, *out_proj;
        bool is_attn;
    } L[6];

    int idx = 0;
    const float* x = (const float*)d_in[idx++];
    for (int i = 0; i < 6; i++) {
        L[i].ln1  = (const float*)d_in[idx++];
        L[i].ln2  = (const float*)d_in[idx++];
        L[i].gate = (const float*)d_in[idx++];
        L[i].up   = (const float*)d_in[idx++];
        L[i].down = (const float*)d_in[idx++];
        L[i].is_attn = (i == 4);
        if (L[i].is_attn) {
            L[i].wq = (const float*)d_in[idx++];
            L[i].wk = (const float*)d_in[idx++];
            L[i].wv = (const float*)d_in[idx++];
            L[i].wo = (const float*)d_in[idx++];
        } else {
            L[i].in_proj  = (const float*)d_in[idx++];
            L[i].conv_w   = (const float*)d_in[idx++];
            L[i].conv_b   = (const float*)d_in[idx++];
            L[i].x_proj   = (const float*)d_in[idx++];
            L[i].dt_ln    = (const float*)d_in[idx++];
            L[i].b_ln     = (const float*)d_in[idx++];
            L[i].c_ln     = (const float*)d_in[idx++];
            L[i].dt_w     = (const float*)d_in[idx++];
            L[i].dt_b     = (const float*)d_in[idx++];
            L[i].A_log    = (const float*)d_in[idx++];
            L[i].Dp       = (const float*)d_in[idx++];
            L[i].out_proj = (const float*)d_in[idx++];
        }
    }
    const float* final_ln = (const float*)d_in[idx++];
    const float* lm_w     = (const float*)d_in[idx++];
    const float* lm_b     = (const float*)d_in[idx++];

    // workspace carve-up (floats)
    float* f = (float*)d_ws;
    size_t o = 0;
    float* h    = f + o; o += (size_t)M * D;        // residual stream
    float* xn   = f + o; o += (size_t)M * D;        // rmsnorm output
    float* big1 = f + o; o += (size_t)M * DFF;      // xz / gate / attn ctx
    float* big2 = f + o; o += (size_t)M * DFF;      // up
    float* ubuf = f + o; o += (size_t)M * DI;       // conv(u) / Q
    float* dtb  = f + o; o += (size_t)M * DI;       // dt / K
    float* ybuf = f + o; o += (size_t)M * DI;       // scan out / V
    float* ssmb = f + o; o += (size_t)M * 96;       // x_proj out
    float* sbuf = f + o; o += (size_t)Bb * H * Ls * Ls;  // attention scores/probs

    (void)hipMemcpyAsync(h, x, (size_t)M * D * sizeof(float),
                         hipMemcpyDeviceToDevice, stream);

    for (int i = 0; i < 6; i++) {
        // ---- mixer branch: h += mix(rmsnorm(h, ln1)) ----
        rmsnorm_kernel<<<M, 256, 0, stream>>>(h, L[i].ln1, xn, D);
        if (L[i].is_attn) {
            launch_gemm(xn, D, L[i].wq, D, nullptr, nullptr, ubuf, D, M, D, D, 0, stream);
            launch_gemm(xn, D, L[i].wk, D, nullptr, nullptr, dtb,  D, M, D, D, 0, stream);
            launch_gemm(xn, D, L[i].wv, D, nullptr, nullptr, ybuf, D, M, D, D, 0, stream);
            // S = QK^T/sqrt(d) with causal mask  (WMMA)
            {
                dim3 g(Ls / 128, Ls / 64, Bb * H);
                attn_score_kernel<<<g, 256, 0, stream>>>(ubuf, dtb, sbuf, Ls);
            }
            softmax_row_kernel<<<Bb * H * Ls, 256, 0, stream>>>(sbuf, Ls);
            // ctx[b,:,h*128:(h+1)*128] = P[b,h] @ V[b,:,h*128:(h+1)*128]  (WMMA)
            for (int bh = 0; bh < Bb * H; bh++) {
                int b  = bh >> 3;
                int hh = bh & 7;
                launch_gemm(sbuf + (size_t)bh * Ls * Ls, Ls,
                            ybuf + (size_t)b * Ls * D + hh * HD, D,
                            nullptr, nullptr,
                            big1 + (size_t)b * Ls * D + hh * HD, D,
                            Ls, HD, Ls, 0, stream);
            }
            launch_gemm(big1, D, L[i].wo, D, nullptr, h, h, D, M, D, D, 0, stream);
        } else {
            // xz = xn @ in_proj   [M, 4096]
            launch_gemm(xn, D, L[i].in_proj, 2 * DI, nullptr, nullptr, big1, 2 * DI, M, 2 * DI, D, 0, stream);
            // u = silu(conv(xz[:, :2048]))
            {
                long long n = (long long)M * DI;
                conv_silu_kernel<<<(unsigned)((n + 255) / 256), 256, 0, stream>>>(
                    big1, L[i].conv_w, L[i].conv_b, ubuf, Ls, M);
            }
            // ssm = u @ x_proj    [M, 96]
            launch_gemm(ubuf, DI, L[i].x_proj, 96, nullptr, nullptr, ssmb, 96, M, 96, DI, 0, stream);
            ssm_norm_kernel<<<(M + 255) / 256, 256, 0, stream>>>(
                ssmb, L[i].dt_ln, L[i].b_ln, L[i].c_ln, M);
            // dt = softplus(ssm[:, :64] @ dt_w + dt_b)   [M, 2048]
            launch_gemm(ssmb, 96, L[i].dt_w, DI, L[i].dt_b, nullptr, dtb, DI, M, DI, 64, 2, stream);
            // selective scan + gating
            mamba_scan_kernel<<<Bb * (DI / 256), 256, 0, stream>>>(
                dtb, ssmb, ubuf, big1, L[i].A_log, L[i].Dp, ybuf, Ls);
            // h += y @ out_proj
            launch_gemm(ybuf, DI, L[i].out_proj, D, nullptr, h, h, D, M, D, DI, 0, stream);
        }

        // ---- MLP branch: h += mlp(rmsnorm(h, ln2)) ----
        rmsnorm_kernel<<<M, 256, 0, stream>>>(h, L[i].ln2, xn, D);
        launch_gemm(xn, D, L[i].gate, DFF, nullptr, nullptr, big1, DFF, M, DFF, D, 1, stream); // silu
        launch_gemm(xn, D, L[i].up,   DFF, nullptr, nullptr, big2, DFF, M, DFF, D, 0, stream);
        {
            long long n = (long long)M * DFF;
            mul_kernel<<<(unsigned)((n + 255) / 256), 256, 0, stream>>>(big1, big2, n);
        }
        launch_gemm(big1, DFF, L[i].down, D, nullptr, h, h, D, M, D, DFF, 0, stream);
    }

    // ---- final norm + LM head ----
    rmsnorm_kernel<<<M, 256, 0, stream>>>(h, final_ln, xn, D);
    launch_gemm(xn, D, lm_w, NV, lm_b, nullptr, (float*)d_out, NV, M, NV, D, 0, stream);
}
